// GroupQueryAttention_40295383171286
// MI455X (gfx1250) — compile-verified, matrix operand order checked by reading, not run on hardware
//
#include <hip/hip_runtime.h>
#include <hip/hip_bf16.h>
#include <stdint.h>

// ---------------- problem constants ----------------
#define BB    2
#define SS    2048
#define HID_  2048
#define NH    32
#define NKV   8
#define HD    64
#define MROWS (BB*SS)          // 4096 tokens
#define SCALE_ 0.125f          // 1/sqrt(64)

typedef __attribute__((ext_vector_type(16))) _Float16 v16h;
typedef __attribute__((ext_vector_type(8)))  _Float16 v8h;
typedef __attribute__((ext_vector_type(8)))  float    v8f;
typedef __attribute__((ext_vector_type(4)))  unsigned int v4u;
typedef __attribute__((ext_vector_type(8)))  unsigned int v8u;

static __device__ __forceinline__ v16h cat16(v8h a, v8h b) {
  v16h r;
#pragma unroll
  for (int i = 0; i < 8; ++i) { r[i] = a[i]; r[i + 8] = b[i]; }
  return r;
}

static __device__ __forceinline__ v8f zero8f() {
  v8f z = {0.f,0.f,0.f,0.f,0.f,0.f,0.f,0.f};
  return z;
}

// ---------------------------------------------------------------------
// Tensor Data Mover: 2-D f16 tile Global -> LDS (CDNA5 ISA D# layout)
//   tile_d0 elements contiguous, tile_d1 rows, row stride `stride` elements.
// Issued once per wave (EXEC ignored by TDM); tracked by TENSORcnt.
// ---------------------------------------------------------------------
static __device__ __forceinline__ void tdm_load_2d(
    unsigned lds_off, const _Float16* gptr,
    unsigned tile_d0, unsigned tile_d1,
    unsigned tensor_d0, unsigned stride) {
  unsigned long long ga = (unsigned long long)(uintptr_t)gptr;
  v4u g0;
  g0[0] = 1u;                                   // count=1, user mode, no gather
  g0[1] = lds_off;                              // lds_addr (bytes)
  g0[2] = (unsigned)(ga & 0xffffffffu);         // global_addr[31:0]
  g0[3] = (unsigned)((ga >> 32) & 0x01ffffffu)  // global_addr[56:32]
          | (2u << 30);                         // type = 2 ("image")
  v8u g1;
  g1[0] = (1u << 16);                           // wg_mask=0, data_size=1 (2B)
  g1[1] = (tensor_d0 & 0xffffu) << 16;          // tensor_dim0[15:0]
  g1[2] = (tensor_d0 >> 16) | (0x0000u << 16);  // tensor_dim0[31:16], tdim1 lo=0x0000
  g1[2] |= (0x0u << 16);                        //   (tensor_dim1 = 0x10000)
  g1[3] = 0x1u | (tile_d0 << 16);               // tensor_dim1[31:16]=1, tile_dim0
  g1[4] = tile_d1;                              // tile_dim1 (tile_dim2 = 0)
  g1[5] = stride;                               // tensor_dim0_stride[31:0]
  g1[6] = 0;                                    // stride[47:32]=0, tdim1_stride lo=0
  g1[7] = 0;                                    // tensor_dim1_stride hi
  v4u g2 = {0u, 0u, 0u, 0u};                    // tensor_dim2.. (unused, 2-D tile)
  v4u g3 = {0u, 0u, 0u, 0u};
  asm volatile("tensor_load_to_lds %0, %1, %2, %3"
               :: "s"(g0), "s"(g1), "s"(g2), "s"(g3)
               : "memory");
}

// =====================================================================
// Generic WMMA GEMM:  D[m][n] = sum_k Amat[m][k] * Bmat[n][k]
//   Block tile: 64 A-rows x 128 B-rows, 8 waves of 32x32, K-chunks of 32.
// MODE 0 (projections): Amat = weight (features), Bmat = activations (tokens)
//   -> lane holds one token, 8 consecutive features: store out[tok][feat] (OT)
// MODE 1 (V, transposed): Amat = activations (tokens), Bmat = wv (features)
//   -> lane holds one feature, 8 consecutive tokens: store Vt[(b*512+f)][s] f16
// =====================================================================
template <typename AT, typename BT, int MODE, typename OT>
__global__ __launch_bounds__(256)
void gemm_nt(const AT* __restrict__ Amat, const BT* __restrict__ Bmat,
             OT* __restrict__ outp, int Arows, int Brows, int Kdim) {
  __shared__ _Float16 lA[64][40];    // padded stride: 80B keeps v8h 16B-aligned
  __shared__ _Float16 lB[128][40];

  const int tid  = threadIdx.x;
  const int lane = tid & 31;
  const int wid  = tid >> 5;
  const int wm   = wid & 1;          // wave M-tile (0..1)
  const int wn   = wid >> 1;         // wave N-tile (0..3)
  const int hi   = lane >> 4;        // lane half
  const int ln   = lane & 15;
  const int aBase = blockIdx.x * 64;
  const int bBase = blockIdx.y * 128;

  v8f acc[2][2];
#pragma unroll
  for (int i = 0; i < 2; ++i)
#pragma unroll
    for (int j = 0; j < 2; ++j) acc[i][j] = zero8f();

  for (int k0 = 0; k0 < Kdim; k0 += 32) {
    { // stage A tile (64x32), convert to f16
      int r = tid >> 2, c0 = (tid & 3) * 8;
      const AT* src = Amat + (size_t)(aBase + r) * Kdim + k0 + c0;
#pragma unroll
      for (int j = 0; j < 8; ++j) lA[r][c0 + j] = (_Float16)src[j];
    }
    { // stage B tile (128x32)
      int r = tid >> 1, c0 = (tid & 1) * 16;
      const BT* src = Bmat + (size_t)(bBase + r) * Kdim + k0 + c0;
#pragma unroll
      for (int j = 0; j < 16; ++j) lB[r][c0 + j] = (_Float16)src[j];
    }
    if (k0 + 32 < Kdim) {  // -> global_prefetch_b8 for next K tile
      __builtin_prefetch(Amat + (size_t)(aBase + (tid >> 2)) * Kdim + k0 + 32, 0, 1);
      __builtin_prefetch(Bmat + (size_t)(bBase + (tid >> 1)) * Kdim + k0 + 32, 0, 1);
    }
    __syncthreads();

    v16h af[2], bf[2];
    const int kb = hi * 8;           // lane-half K split: 0..7 / 8..15 (+16)
#pragma unroll
    for (int ms = 0; ms < 2; ++ms) {
      int r = wm * 32 + ms * 16 + ln;
      af[ms] = cat16(*(const v8h*)&lA[r][kb], *(const v8h*)&lA[r][16 + kb]);
    }
#pragma unroll
    for (int ns = 0; ns < 2; ++ns) {
      int r = wn * 32 + ns * 16 + ln;
      bf[ns] = cat16(*(const v8h*)&lB[r][kb], *(const v8h*)&lB[r][16 + kb]);
    }
#pragma unroll
    for (int ms = 0; ms < 2; ++ms)
#pragma unroll
      for (int ns = 0; ns < 2; ++ns)
        acc[ms][ns] = __builtin_amdgcn_wmma_f32_16x16x32_f16(
            false, af[ms], false, bf[ns], (short)0, acc[ms][ns], false, false);
    __syncthreads();
  }

  // -------- store (C layout: lane = N col, VGPR i = M row i (+8 for hi)) ----
#pragma unroll
  for (int ms = 0; ms < 2; ++ms)
#pragma unroll
    for (int ns = 0; ns < 2; ++ns) {
      if constexpr (MODE == 0) {
        int t     = bBase + wn * 32 + ns * 16 + ln;       // token (N)
        int fbase = aBase + wm * 32 + ms * 16 + hi * 8;   // feature (M)
        if constexpr (sizeof(OT) == 2) {
          v8h v;
#pragma unroll
          for (int i = 0; i < 8; ++i) v[i] = (_Float16)acc[ms][ns][i];
          *(v8h*)(outp + (size_t)t * Arows + fbase) = v;
        } else {
          float* po = (float*)(outp + (size_t)t * Arows + fbase);
          float4 v0 = make_float4(acc[ms][ns][0], acc[ms][ns][1], acc[ms][ns][2], acc[ms][ns][3]);
          float4 v1 = make_float4(acc[ms][ns][4], acc[ms][ns][5], acc[ms][ns][6], acc[ms][ns][7]);
          *(float4*)po       = v0;
          *(float4*)(po + 4) = v1;
        }
      } else {  // MODE 1: Vt[(batch*512 + feature)][token]
        int tbase = aBase + wm * 32 + ms * 16 + hi * 8;   // token (M)
        int f     = bBase + wn * 32 + ns * 16 + ln;       // feature (N)
        int bb = tbase >> 11, s0 = tbase & (SS - 1);
        v8h v;
#pragma unroll
        for (int i = 0; i < 8; ++i) v[i] = (_Float16)acc[ms][ns][i];
        *(v8h*)((_Float16*)outp + ((size_t)(bb * (NKV * HD) + f)) * SS + s0) = v;
      }
    }
}

// =====================================================================
// RoPE (in place, f16):  x is [MROWS][nheads*64];  pairs (d, d+32)
// =====================================================================
__global__ void rope_kernel(_Float16* __restrict__ x, int nheads) {
  int idx = blockIdx.x * blockDim.x + threadIdx.x;
  int total = MROWS * nheads * 32;
  if (idx >= total) return;
  int d = idx & 31;
  int h = (idx >> 5) % nheads;
  int m = idx / (32 * nheads);
  int s = m & (SS - 1);
  float inv = __powf(10000.0f, -(float)d * (1.0f / 32.0f));
  float fr = (float)s * inv;
  float sn, cs;
  __sincosf(fr, &sn, &cs);
  size_t base = (size_t)m * (nheads * HD) + h * HD + d;
  float x0 = (float)x[base], x1 = (float)x[base + 32];
  x[base]      = (_Float16)(x0 * cs - x1 * sn);
  x[base + 32] = (_Float16)(x1 * cs + x0 * sn);
}

// =====================================================================
// Flash attention: one (b, head, 64-query tile) per 4-wave block.
// Wave 0 drives a double-buffered TDM pipeline staging K (32x64) and
// V^T (64x32) tiles into LDS; all waves consume them via ds_load_b128.
// Each wave owns 16 queries x full head_dim with online softmax.
// =====================================================================
__global__ __launch_bounds__(128)
void attn_kernel(const _Float16* __restrict__ Q, const _Float16* __restrict__ Km,
                 const _Float16* __restrict__ Vt, _Float16* __restrict__ AO) {
  __shared__ _Float16 lK[2][32][64];   // [buf][key][d]      8 KB
  __shared__ _Float16 lV[2][64][32];   // [buf][d][key]      8 KB
  __shared__ _Float16 lp[4][16][40];   // per-wave P staging 5 KB

  const int lane = threadIdx.x & 31;
  const int w    = threadIdx.x >> 5;
  const int hi   = lane >> 4;
  const int ln   = lane & 15;
  const int qt   = blockIdx.x;
  const int h    = blockIdx.y;
  const int b    = blockIdx.z;
  const int kv   = h >> 2;             // GROUP = 4
  const int q0   = qt * 64 + w * 16;
  const int kb   = hi * 8;

  // Q A-fragments for head-dim chunks [0,32) and [32,64)
  const _Float16* qrow = Q + ((size_t)(b * SS + q0 + ln)) * HID_ + h * HD;
  v16h qf[2];
#pragma unroll
  for (int kc = 0; kc < 2; ++kc)
    qf[kc] = cat16(*(const v8h*)(qrow + kc * 32 + kb),
                   *(const v8h*)(qrow + kc * 32 + 16 + kb));

  v8f o[4];
#pragma unroll
  for (int nt = 0; nt < 4; ++nt) o[nt] = zero8f();
  float rmax[8], rsum[8];
#pragma unroll
  for (int i = 0; i < 8; ++i) { rmax[i] = -3.0e38f; rsum[i] = 0.0f; }

  const _Float16* kbase = Km + (size_t)(b * SS) * (NKV * HD) + kv * HD;      // +kt*512
  const _Float16* vbase = Vt + (size_t)(b * (NKV * HD) + kv * HD) * SS;      // +d*SS+kt
  const int niter = (qt * 64 + 64) >> 5;

  if (w == 0) {  // prime the pipeline: tile 0 -> buffer 0
    tdm_load_2d((unsigned)(uintptr_t)&lK[0][0][0], kbase, HD, 32, HD, NKV * HD);
    tdm_load_2d((unsigned)(uintptr_t)&lV[0][0][0], vbase, 32, HD, 32, SS);
  }

  for (int it = 0; it < niter; ++it) {
    const int kt  = it << 5;
    const int buf = it & 1;
    if (w == 0) {
      if (it + 1 < niter) {  // issue next tile, then drain current pair
        tdm_load_2d((unsigned)(uintptr_t)&lK[buf ^ 1][0][0],
                    kbase + (size_t)(kt + 32) * (NKV * HD), HD, 32, HD, NKV * HD);
        tdm_load_2d((unsigned)(uintptr_t)&lV[buf ^ 1][0][0],
                    vbase + kt + 32, 32, HD, 32, SS);
        __builtin_amdgcn_s_wait_tensorcnt(2);
      } else {
        __builtin_amdgcn_s_wait_tensorcnt(0);
      }
    }
    __syncthreads();

    // ---- scores S = Q K^T from LDS K tile ----
    v8f sc[2];
#pragma unroll
    for (int ns = 0; ns < 2; ++ns) {
      const _Float16* kr = &lK[buf][ns * 16 + ln][0];
      v16h kf0 = cat16(*(const v8h*)(kr + kb),      *(const v8h*)(kr + 16 + kb));
      v16h kf1 = cat16(*(const v8h*)(kr + 32 + kb), *(const v8h*)(kr + 48 + kb));
      sc[ns] = __builtin_amdgcn_wmma_f32_16x16x32_f16(false, qf[0], false, kf0,
                                                      (short)0, zero8f(), false, false);
      sc[ns] = __builtin_amdgcn_wmma_f32_16x16x32_f16(false, qf[1], false, kf1,
                                                      (short)0, sc[ns], false, false);
    }

    // ---- scale (+ causal mask only on diagonal tiles) ----
    float p0[8], p1[8], tmax[8];
#pragma unroll
    for (int i = 0; i < 8; ++i) { p0[i] = sc[0][i] * SCALE_; p1[i] = sc[1][i] * SCALE_; }
    if (kt + 31 > q0) {   // wave-uniform: skipped for fully-causal-legal tiles
#pragma unroll
      for (int i = 0; i < 8; ++i) {
        int qr = q0 + i + hi * 8;
        p0[i] += ((kt + ln)      > qr ? -1000000000.0f : 0.f);
        p1[i] += ((kt + 16 + ln) > qr ? -1000000000.0f : 0.f);
      }
    }
    // ---- online softmax (row = lane-half, reduce across 16 lanes) ----
#pragma unroll
    for (int i = 0; i < 8; ++i) tmax[i] = fmaxf(p0[i], p1[i]);
#pragma unroll
    for (int off = 1; off < 16; off <<= 1)
#pragma unroll
      for (int i = 0; i < 8; ++i)
        tmax[i] = fmaxf(tmax[i], __shfl_xor(tmax[i], off, 16));
    float alpha[8];
#pragma unroll
    for (int i = 0; i < 8; ++i) {
      float nm = fmaxf(rmax[i], tmax[i]);
      alpha[i] = __expf(rmax[i] - nm);
      rmax[i] = nm;
      p0[i] = __expf(p0[i] - nm);
      p1[i] = __expf(p1[i] - nm);
    }
    float psum[8];
#pragma unroll
    for (int i = 0; i < 8; ++i) psum[i] = p0[i] + p1[i];
#pragma unroll
    for (int off = 1; off < 16; off <<= 1)
#pragma unroll
      for (int i = 0; i < 8; ++i)
        psum[i] += __shfl_xor(psum[i], off, 16);
#pragma unroll
    for (int i = 0; i < 8; ++i) rsum[i] = rsum[i] * alpha[i] + psum[i];
#pragma unroll
    for (int nt = 0; nt < 4; ++nt)
#pragma unroll
      for (int i = 0; i < 8; ++i) o[nt][i] *= alpha[i];

    // ---- transpose P: C-layout -> A-layout via per-wave LDS region ----
#pragma unroll
    for (int i = 0; i < 8; ++i) {
      lp[w][i + hi * 8][ln]      = (_Float16)p0[i];
      lp[w][i + hi * 8][16 + ln] = (_Float16)p1[i];
    }
    // per-wave LDS RAW fence (LDS is in-order per wave; this pins the compiler)
    asm volatile("s_wait_dscnt 0x0" ::: "memory");
    v16h pa = cat16(*(const v8h*)&lp[w][ln][kb],
                    *(const v8h*)&lp[w][ln][16 + kb]);

    // ---- O += P * V from LDS V^T tile ----
#pragma unroll
    for (int nt = 0; nt < 4; ++nt) {
      const _Float16* vr = &lV[buf][nt * 16 + ln][0];
      v16h vf = cat16(*(const v8h*)(vr + kb), *(const v8h*)(vr + 16 + kb));
      o[nt] = __builtin_amdgcn_wmma_f32_16x16x32_f16(false, pa, false, vf,
                                                     (short)0, o[nt], false, false);
    }
    __syncthreads();   // all readers done before buf is overwritten next iter
  }

  // ---- normalize and store attn output [b,s,h*64+d] ----
  float inv[8];
#pragma unroll
  for (int i = 0; i < 8; ++i) inv[i] = 1.0f / rsum[i];
#pragma unroll
  for (int nt = 0; nt < 4; ++nt)
#pragma unroll
    for (int i = 0; i < 8; ++i) {
      int qr = q0 + i + hi * 8;
      AO[((size_t)(b * SS + qr)) * HID_ + h * HD + nt * 16 + ln] =
          (_Float16)(o[nt][i] * inv[i]);
    }
}

// =====================================================================
extern "C" void kernel_launch(void* const* d_in, const int* in_sizes, int n_in,
                              void* d_out, int out_size, void* d_ws, size_t ws_size,
                              hipStream_t stream) {
  const float* hidden = (const float*)d_in[0];
  // d_in[1] = attn_mask: pure causal triu(-1e9); computed analytically in-kernel
  const float* wq = (const float*)d_in[2];
  const float* wk = (const float*)d_in[3];
  const float* wv = (const float*)d_in[4];
  const float* wo = (const float*)d_in[5];
  float* out = (float*)d_out;

  char* ws = (char*)d_ws;
  const size_t qBytes = (size_t)MROWS * HID_ * 2;        // 16 MiB
  const size_t kBytes = (size_t)MROWS * (NKV * HD) * 2;  // 4 MiB
  _Float16* Qh = (_Float16*)(ws);
  _Float16* Kh = (_Float16*)(ws + qBytes);
  _Float16* Vt = (_Float16*)(ws + qBytes + kBytes);
  _Float16* AO = (_Float16*)(ws + qBytes + 2 * kBytes);

  dim3 blk(256);
  // Q = hidden @ wq^T  (stored [tok][2048] f16)
  gemm_nt<float, float, 0, _Float16>
      <<<dim3(HID_ / 64, MROWS / 128), blk, 0, stream>>>(wq, hidden, Qh, HID_, MROWS, HID_);
  // K = hidden @ wk^T  (stored [tok][512] f16)
  gemm_nt<float, float, 0, _Float16>
      <<<dim3((NKV * HD) / 64, MROWS / 128), blk, 0, stream>>>(wk, hidden, Kh, NKV * HD, MROWS, HID_);
  // V = hidden @ wv^T  (stored transposed: Vt[(b*512+f)][s] f16)
  gemm_nt<float, float, 1, _Float16>
      <<<dim3(MROWS / 64, (NKV * HD) / 128), blk, 0, stream>>>(hidden, wv, Vt, MROWS, NKV * HD, HID_);
  // RoPE on Q and K
  rope_kernel<<<(MROWS * NH * 32) / 256, 256, 0, stream>>>(Qh, NH);
  rope_kernel<<<(MROWS * NKV * 32) / 256, 256, 0, stream>>>(Kh, NKV);
  // Flash attention with TDM-staged K/V tiles
  attn_kernel<<<dim3(SS / 64, NH, BB), dim3(128), 0, stream>>>(Qh, Kh, Vt, AO);
  // out = AO @ wo^T (f32 straight to d_out)
  gemm_nt<float, _Float16, 0, float>
      <<<dim3(HID_ / 64, MROWS / 128), blk, 0, stream>>>(wo, AO, out, HID_, MROWS, HID_);
}
